// KANLayer_75118978007752
// MI455X (gfx1250) — matrix operand order repaired and checked by reading, not compiled, not run on hardware
//
#include <hip/hip_runtime.h>

// KAN layer on MI455X (gfx1250, wave32, WMMA).
// out[b,o] = sum_i [ sum_n bases_n(x[b,i]) * coeff[o,i,n] * spline_w[o,i]
//                    + silu(x[b,i]) * base_w[o,i] ]
// Recast as GEMM: A[1024 x 2304] (8 spline bases + silu per input feature)
// times W^T[2304 x 256] (fused coeff*spline_w + base_w), K = 256*9 = 72*32.
// f16 storage / f32 WMMA accumulation via v_wmma_f32_16x16x32_f16.

typedef __attribute__((ext_vector_type(16))) _Float16 v16h;
typedef __attribute__((ext_vector_type(8)))  float    v8f;
typedef __attribute__((ext_vector_type(4)))  float    f4;   // 16-byte chunk

#define B_SZ   1024
#define IN_F   256
#define OUT_F  256
#define NBAS   9                     // 8 spline bases + 1 silu channel
#define KDIM   (IN_F * NBAS)         // 2304 = 72 * 32

// ---------------------------------------------------------------------------
// Stage 0: fuse weights into W[o][k], k = i*9+n, K-contiguous f16 rows.
// ---------------------------------------------------------------------------
__global__ void kan_pack_w(const float* __restrict__ coeff,
                           const float* __restrict__ base_w,
                           const float* __restrict__ spline_w,
                           _Float16* __restrict__ Wp) {
  int idx = blockIdx.x * blockDim.x + threadIdx.x;
  if (idx >= OUT_F * KDIM) return;
  int o = idx / KDIM;
  int k = idx - o * KDIM;
  int i = k / NBAS;
  int n = k - i * NBAS;
  float v;
  if (n < 8) v = coeff[(o * IN_F + i) * 8 + n] * spline_w[o * IN_F + i];
  else       v = base_w[o * IN_F + i];
  Wp[idx] = (_Float16)v;
}

// ---------------------------------------------------------------------------
// Stage 1: cubic B-spline bases (Cox-de Boor, clipped denominators like the
// reference) + last-basis correction + silu -> A[b][i*9+n] f16.
// ---------------------------------------------------------------------------
__global__ void kan_expand(const float* __restrict__ x,
                           const float* __restrict__ grid,
                           _Float16* __restrict__ Ap) {
  int idx = blockIdx.x * blockDim.x + threadIdx.x;
  if (idx >= B_SZ * IN_F) return;
  float xv = x[idx];

  float g[12];
#pragma unroll
  for (int j = 0; j < 12; ++j) g[j] = grid[j];

  float bas[11];
#pragma unroll
  for (int j = 0; j < 11; ++j)
    bas[j] = (xv >= g[j] && xv < g[j + 1]) ? 1.0f : 0.0f;

#pragma unroll
  for (int p = 1; p <= 3; ++p) {
#pragma unroll
    for (int j = 0; j < 11 - p; ++j) {
      float ld = fmaxf(g[j + p] - g[j], 1e-8f);
      float rd = fmaxf(g[j + p + 1] - g[j + 1], 1e-8f);
      bas[j] = (xv - g[j]) / ld * bas[j] + (g[j + p + 1] - xv) / rd * bas[j + 1];
    }
  }
  bas[7] += (xv >= g[8]) ? 1.0f : 0.0f;      // grid[-K-1] correction
  float sil = xv / (1.0f + __expf(-xv));     // silu(x)

  _Float16* dst = Ap + (size_t)idx * NBAS;
#pragma unroll
  for (int n = 0; n < 8; ++n) dst[n] = (_Float16)bas[n];
  dst[8] = (_Float16)sil;
}

// ---------------------------------------------------------------------------
// Stage 2: out = A * W^T with v_wmma_f32_16x16x32_f16.
// One 16(M) x 32(N) tile per wave; A fragment reused across 2 WMMAs.
//
// Per-lane fragment addressing (CDNA5 ISA 7.12.2, wave32):
//   A 16x32 f16:  row = lane&15; lanes 0-15 hold K {0..7,16..23},
//                 lanes 16-31 hold K {8..15,24..31}  -> two b128 loads.
//   B 32x16 f16:  col = lane&15; lanes 0-15 hold K {0..7,16..23},
//                 lanes 16-31 hold K {8..15,24..31} (K-contiguous in W rows)
//                 -> two b128 loads.
//   D 16x16 f32:  VGPR r -> row r + 8*(lane>=16), col = lane&15.
// ---------------------------------------------------------------------------
__global__ void __launch_bounds__(256) kan_gemm(const _Float16* __restrict__ Ap,
                                                const _Float16* __restrict__ Wp,
                                                float* __restrict__ out) {
  const int lane = threadIdx.x & 31;
  const int wave = threadIdx.x >> 5;     // 8 waves per block
  const int half = lane >> 4;            // 0 | 1
  const int l15  = lane & 15;
  const int m_base = (blockIdx.x * 8 + wave) * 16;   // 128 rows per block
  const int n_base = blockIdx.y * 32;                // 32 cols per wave

  const _Float16* pa  = Ap + (size_t)(m_base + l15) * KDIM + half * 8;
  const _Float16* pb0 = Wp + (size_t)(n_base + l15) * KDIM + half * 8;
  const _Float16* pb1 = pb0 + (size_t)16 * KDIM;

  v8f acc0 = {};
  v8f acc1 = {};

  for (int k = 0; k < KDIM; k += 32) {
    v16h a, b0, b1;
    ((f4*)&a)[0]  = *(const f4*)(pa  + k);        // K koff+0..7
    ((f4*)&a)[1]  = *(const f4*)(pa  + k + 16);   // K koff+16..23
    ((f4*)&b0)[0] = *(const f4*)(pb0 + k);        // K koff+0..7
    ((f4*)&b0)[1] = *(const f4*)(pb0 + k + 16);   // K koff+16..23
    ((f4*)&b1)[0] = *(const f4*)(pb1 + k);
    ((f4*)&b1)[1] = *(const f4*)(pb1 + k + 16);

    acc0 = __builtin_amdgcn_wmma_f32_16x16x32_f16(
        false, a, false, b0, (short)0, acc0, false, false);
    acc1 = __builtin_amdgcn_wmma_f32_16x16x32_f16(
        false, a, false, b1, (short)0, acc1, false, false);
  }

  float* o0 = out + (size_t)(m_base + half * 8) * OUT_F + n_base + l15;
#pragma unroll
  for (int r = 0; r < 8; ++r) {
    o0[(size_t)r * OUT_F]      = acc0[r];
    o0[(size_t)r * OUT_F + 16] = acc1[r];
  }
}

// ---------------------------------------------------------------------------
extern "C" void kernel_launch(void* const* d_in, const int* in_sizes, int n_in,
                              void* d_out, int out_size, void* d_ws, size_t ws_size,
                              hipStream_t stream) {
  (void)in_sizes; (void)n_in; (void)out_size; (void)ws_size;

  const float* x        = (const float*)d_in[0];   // (1024, 256)
  const float* grid     = (const float*)d_in[1];   // (12,)
  const float* coeff    = (const float*)d_in[2];   // (256, 256, 8)
  const float* base_w   = (const float*)d_in[3];   // (256, 256)
  const float* spline_w = (const float*)d_in[4];   // (256, 256)
  float* out = (float*)d_out;                      // (1024, 256)

  _Float16* Wp = (_Float16*)d_ws;                                  // 1.18 MB
  _Float16* Ap = (_Float16*)((char*)d_ws +
                             (size_t)OUT_F * KDIM * sizeof(_Float16)); // 4.72 MB

  kan_pack_w<<<(OUT_F * KDIM + 255) / 256, 256, 0, stream>>>(
      coeff, base_w, spline_w, Wp);
  kan_expand<<<(B_SZ * IN_F + 255) / 256, 256, 0, stream>>>(
      x, grid, Ap);

  dim3 gemm_grid(B_SZ / 128, OUT_F / 32);   // 8 x 8 blocks, 8 waves each
  kan_gemm<<<gemm_grid, 256, 0, stream>>>(Ap, Wp, out);
}
